// EdgeUpdate_2860448219508
// MI455X (gfx1250) — matrix operand order repaired and checked by reading, not compiled, not run on hardware
//
#include <hip/hip_runtime.h>
#include <math.h>

// ---------------------------------------------------------------------------
// EdgeUpdate (CGCNN/MEGNet-style), MI455X / gfx1250, wave32 + WMMA f16->f32
//
// Factorization: h3 rows = PnA[b,a] + m_ij*PnB[b,jj] + PnC[b,kk] + PeD[b,a,j]
//                          + PeE[b,jj,k] + b3      (five K=128 projections)
// GEMM tiles stage f32 into LDS via GLOBAL_LOAD_ASYNC_TO_LDS (ASYNCcnt);
// f32->f16 conversion happens during WMMA fragment assembly.
// ---------------------------------------------------------------------------

#define B_   4
#define AT_  64
#define NBR_ 24
#define FN_  128
#define FE_  128
#define ROWS2 (B_*AT_*NBR_)          // 6144  (b,a,j) rows
#define NODES (B_*AT_)               // 256

typedef _Float16 v16h __attribute__((ext_vector_type(16)));
typedef float    v8f  __attribute__((ext_vector_type(8)));

// f32 LDS tile strides (dwords), padded for conflict-free WMMA fragment reads:
// A stride 132 -> per-row bank step 4; W^T stride 36 -> bank starts 36m mod 64
// all distinct (gap >= 4), each ds_load_b128 touches 4 banks.
#define AT_STRIDE 132
#define WT_STRIDE 36

// ---- gfx1250 async-to-LDS path (guarded; falls back cleanly) ---------------
#if defined(__has_builtin)
#if __has_builtin(__builtin_amdgcn_global_load_async_to_lds_b32) && \
    __has_builtin(__builtin_amdgcn_global_load_async_to_lds_b128)
#define HAVE_ASYNC_LDS 1
#endif
#endif

#ifdef HAVE_ASYNC_LDS
// Builtin signatures (from clang diagnostics):
//   b32 : (addrspace(1) int*, addrspace(3) int*, imm int, imm int)
//   b128: (addrspace(1) v4i*, addrspace(3) v4i*, imm int, imm int)
typedef int v4i_b __attribute__((vector_size(16)));
#define GAS32(p)  ((__attribute__((address_space(1))) int*)(p))
#define LAS32(p)  ((__attribute__((address_space(3))) int*)(p))
#define GAS128(p) ((__attribute__((address_space(1))) v4i_b*)(p))
#define LAS128(p) ((__attribute__((address_space(3))) v4i_b*)(p))
__device__ __forceinline__ void wait_async0() {
#if __has_builtin(__builtin_amdgcn_s_wait_asynccnt)
    __builtin_amdgcn_s_wait_asynccnt(0);
#else
    asm volatile("s_wait_asynccnt 0" ::: "memory");
#endif
}
#endif

// Build a 16-half WMMA fragment from two 8-float LDS chunks.
__device__ __forceinline__ v16h cvt_frag(v8f lo, v8f hi) {
    v16h r;
#pragma unroll
    for (int i = 0; i < 8; ++i) { r[i] = (_Float16)lo[i]; r[8 + i] = (_Float16)hi[i]; }
    return r;
}

// A fragment (16x32 f16, MxK), ISA layout: lane m holds row m; elements 0..7
// hold K = kb+8g.., elements 8..15 hold K = kb+16+8g..
__device__ __forceinline__ v16h load_a_frag(const float* At, int lane, int kb) {
    const int m = lane & 15, g = lane >> 4;
    const float* row = At + m * AT_STRIDE;
    return cvt_frag(*(const v8f*)(row + kb + 8 * g),
                    *(const v8f*)(row + kb + 16 + 8 * g));
}

// B fragment (32x16 f16, KxN) from transposed tile Wt[n][k]: lane owns column
// n0+(lane&15), elements 0..7 hold K = 8g.., elements 8..15 hold K = 16+8g..
__device__ __forceinline__ v16h load_b_frag(const float* Wt, int lane, int n0) {
    const int n = n0 + (lane & 15), g = lane >> 4;
    const float* row = Wt + n * WT_STRIDE;
    return cvt_frag(*(const v8f*)(row + 8 * g),
                    *(const v8f*)(row + 16 + 8 * g));
}

// Stage one 32x256 W chunk transposed into Wt[n][k] (f32). tid in [0,256).
__device__ __forceinline__ void stage_w_chunk(float* Wt, const float* W,
                                              int ks, int tid) {
    const float* wsrc = W + (size_t)(ks * 32) * 256 + tid;
#ifdef HAVE_ASYNC_LDS
#pragma unroll
    for (int i = 0; i < 32; ++i)
        __builtin_amdgcn_global_load_async_to_lds_b32(
            GAS32(wsrc + (size_t)i * 256), LAS32(Wt + tid * WT_STRIDE + i), 0, 0);
#else
#pragma unroll
    for (int i = 0; i < 32; ++i)
        Wt[tid * WT_STRIDE + i] = wsrc[(size_t)i * 256];
#endif
}

__device__ __forceinline__ void stage_fence() {
#ifdef HAVE_ASYNC_LDS
    wait_async0();            // this wave's async copies landed in LDS
#endif
    __syncthreads();          // every wave's copies landed
}

// ---------------------------------------------------------------------------
// Projection GEMM: Out[M,256] = A[M,128] @ W[128,256]   (f16 WMMA, f32 acc)
// Block = 256 threads (8 waves); one 16-row M tile; wave w owns N tiles
// {2w, 2w+1}; K loop: 4 steps of 32.
// ---------------------------------------------------------------------------
__global__ __launch_bounds__(256) void gemm_proj_kernel(
    const float* __restrict__ A, const float* __restrict__ W,
    float* __restrict__ Out)
{
    __shared__ __align__(16) float At[16 * AT_STRIDE];
    __shared__ __align__(16) float Wt[256 * WT_STRIDE];

    const int tid = threadIdx.x, lane = tid & 31, wave = tid >> 5;
    const int mtile = blockIdx.x;

    // Stage A tile (16x128 f32): contiguous 32B per thread.
    {
        const int row = tid & 15, chunk = tid >> 4;
        const float* src = A + (size_t)(mtile * 16 + row) * FN_ + chunk * 8;
        float* dst = At + row * AT_STRIDE + chunk * 8;
#ifdef HAVE_ASYNC_LDS
        __builtin_amdgcn_global_load_async_to_lds_b128(GAS128(src),     LAS128(dst),     0, 0);
        __builtin_amdgcn_global_load_async_to_lds_b128(GAS128(src + 4), LAS128(dst + 4), 0, 0);
#else
#pragma unroll
        for (int i = 0; i < 8; ++i) dst[i] = src[i];
#endif
    }

    v8f c0 = {}, c1 = {};
    const int n0 = wave * 32, n1 = n0 + 16;

    for (int ks = 0; ks < 4; ++ks) {
        __syncthreads();                      // Wt safe to overwrite
        stage_w_chunk(Wt, W, ks, tid);
        stage_fence();

        v16h a  = load_a_frag(At, lane, ks * 32);
        v16h b0 = load_b_frag(Wt, lane, n0);
        v16h b1 = load_b_frag(Wt, lane, n1);
        c0 = __builtin_amdgcn_wmma_f32_16x16x32_f16(false, a, false, b0, (short)0, c0, false, false);
        c1 = __builtin_amdgcn_wmma_f32_16x16x32_f16(false, a, false, b1, (short)0, c1, false, false);
    }

    // C/D layout: row = v + 8g, col = lane&15.
    const int m = lane & 15, g = lane >> 4;
#pragma unroll
    for (int v = 0; v < 8; ++v) {
        const size_t r = (size_t)(mtile * 16 + 8 * g + v) * 256;
        Out[r + n0 + m] = c0[v];
        Out[r + n1 + m] = c1[v];
    }
}

// ---------------------------------------------------------------------------
// Two-body: h2 = (node_i * node_j * mask) @ W2 + b2 ;
// gate = sigmoid(h2[:128]) * tanh(h2[128:]). Wave w owns feature tile
// [16w,16w+16) paired with +128 so the gate combines accumulators in-register.
// ---------------------------------------------------------------------------
__global__ __launch_bounds__(256) void twobody_kernel(
    const float* __restrict__ node, const float* __restrict__ W2,
    const float* __restrict__ b2, const int* __restrict__ nbr_idx,
    const float* __restrict__ nbr_mask, float* __restrict__ two_body)
{
    __shared__ __align__(16) float At[16 * AT_STRIDE];
    __shared__ __align__(16) float Wt[256 * WT_STRIDE];

    const int tid = threadIdx.x, lane = tid & 31, wave = tid >> 5;
    const int mtile = blockIdx.x;              // 384 tiles over 6144 rows

    {   // Stage A: row r=(b,a,j): arow = node[b,a,:] * node[b,jj,:] * mask
        const int row = tid & 15, chunk = tid >> 4;
        const int r  = mtile * 16 + row;
        const int b  = r / (AT_ * NBR_);
        const int a  = (r / NBR_) % AT_;
        const int jj = nbr_idx[r];
        const float mk = nbr_mask[r];
        const float* ni = node + (size_t)(b * AT_ + a)  * FN_ + chunk * 8;
        const float* nj = node + (size_t)(b * AT_ + jj) * FN_ + chunk * 8;
        float* dst = At + row * AT_STRIDE + chunk * 8;
#pragma unroll
        for (int i = 0; i < 8; ++i) dst[i] = ni[i] * nj[i] * mk;
    }

    v8f c0 = {}, c1 = {};
    const int n0 = wave * 16;                  // paired tile is n0+128

    for (int ks = 0; ks < 4; ++ks) {
        __syncthreads();
        stage_w_chunk(Wt, W2, ks, tid);
        stage_fence();

        v16h a  = load_a_frag(At, lane, ks * 32);
        v16h b0 = load_b_frag(Wt, lane, n0);
        v16h b1 = load_b_frag(Wt, lane, n0 + 128);
        c0 = __builtin_amdgcn_wmma_f32_16x16x32_f16(false, a, false, b0, (short)0, c0, false, false);
        c1 = __builtin_amdgcn_wmma_f32_16x16x32_f16(false, a, false, b1, (short)0, c1, false, false);
    }

    const int m = lane & 15, g = lane >> 4;
    const int f = n0 + m;
    const float bb0 = b2[f], bb1 = b2[f + 128];
#pragma unroll
    for (int v = 0; v < 8; ++v) {
        const int r = mtile * 16 + 8 * g + v;
        const float h0 = c0[v] + bb0;
        const float h1 = c1[v] + bb1;
        const float sg = 1.0f / (1.0f + __expf(-h0));
        two_body[(size_t)r * FE_ + f] = sg * tanhf(h1);
    }
}

// ---------------------------------------------------------------------------
// Three-body accumulation over k using precomputed projections.
// One wave per (b,a,j); lane L owns features {L+32i}; gate pairs i with i+4
// (f vs f+128). Table reads are 128B-coalesced; next-k rows are prefetched.
// ---------------------------------------------------------------------------
__global__ __launch_bounds__(256) void threebody_kernel(
    const float* __restrict__ PnA, const float* __restrict__ PnB,
    const float* __restrict__ PnC, const float* __restrict__ PeD,
    const float* __restrict__ PeE, const float* __restrict__ b3,
    const int* __restrict__ nbr_idx, const float* __restrict__ nbr_mask,
    float* __restrict__ three)
{
    const int tid = threadIdx.x, lane = tid & 31, wave = tid >> 5;
    const int r  = blockIdx.x * 8 + wave;      // 768 blocks * 8 waves = 6144
    const int b  = r / (AT_ * NBR_);
    const int ba = r / NBR_;                   // b*AT + a
    const int jj = nbr_idx[r];
    const int bj = b * AT_ + jj;
    const float mij = nbr_mask[r];

    float base[8];
#pragma unroll
    for (int i = 0; i < 8; ++i) {
        const int f = lane + 32 * i;
        base[i] = PnA[(size_t)ba * 256 + f] + mij * PnB[(size_t)bj * 256 + f]
                + PeD[(size_t)r * 256 + f] + b3[f];
    }

    float acc[4] = {0.f, 0.f, 0.f, 0.f};
    for (int k = 0; k < NBR_; ++k) {
        const int   kk = nbr_idx[bj * NBR_ + k];
        const float mk = nbr_mask[bj * NBR_ + k];
        const float* pc = PnC + (size_t)(b * AT_ + kk) * 256;
        const float* pe = PeE + (size_t)(bj * NBR_ + k) * 256;
        if (k + 1 < NBR_) {
            const int kk2 = nbr_idx[bj * NBR_ + k + 1];
            __builtin_prefetch(PnC + (size_t)(b * AT_ + kk2) * 256 + lane, 0, 3);
            __builtin_prefetch(pe + 256 + lane, 0, 3);
        }
        float v[8];
#pragma unroll
        for (int i = 0; i < 8; ++i) {
            const int f = lane + 32 * i;
            v[i] = base[i] + pc[f] + pe[f];
        }
#pragma unroll
        for (int i = 0; i < 4; ++i) {
            const float sg = 1.0f / (1.0f + __expf(-v[i]));
            acc[i] += mk * sg * tanhf(v[i + 4]);
        }
    }
#pragma unroll
    for (int i = 0; i < 4; ++i)
        three[(size_t)r * FE_ + lane + 32 * i] = acc[i];
}

// ---------------------------------------------------------------------------
// BatchNorm statistics + fused epilogue
// ---------------------------------------------------------------------------
__global__ __launch_bounds__(256) void zero_sums_kernel(float* __restrict__ sums) {
    sums[threadIdx.x] = 0.0f;                  // 256 floats: sum | sumsq
}

__global__ __launch_bounds__(256) void bnsum_kernel(
    const float* __restrict__ three, float* __restrict__ sums)
{
    const int f = threadIdx.x & 127;
    const int rstart = blockIdx.x * 2 + (threadIdx.x >> 7);
    float s = 0.f, s2 = 0.f;
    for (int r = rstart; r < ROWS2; r += gridDim.x * 2) {
        const float x = three[(size_t)r * FE_ + f];
        s += x; s2 += x * x;
    }
    atomicAdd(&sums[f], s);
    atomicAdd(&sums[128 + f], s2);
}

__global__ __launch_bounds__(256) void final_kernel(
    const float* __restrict__ edge, const float* __restrict__ two_body,
    const float* __restrict__ three, const float* __restrict__ sums,
    const float* __restrict__ gamma, const float* __restrict__ beta,
    float* __restrict__ out)
{
    const int idx = blockIdx.x * 256 + threadIdx.x;
    if (idx >= ROWS2 * FE_) return;
    const int f = idx & 127;
    const float inv_n = 1.0f / (float)ROWS2;
    const float mean = sums[f] * inv_n;
    const float var  = sums[128 + f] * inv_n - mean * mean;
    const float bn   = (three[idx] - mean) * rsqrtf(var + 1e-5f) * gamma[f] + beta[f];
    out[idx] = tanhf(edge[idx] + two_body[idx] + bn);
}

// ---------------------------------------------------------------------------
// Host launcher
// ---------------------------------------------------------------------------
extern "C" void kernel_launch(void* const* d_in, const int* in_sizes, int n_in,
                              void* d_out, int out_size, void* d_ws, size_t ws_size,
                              hipStream_t stream)
{
    (void)in_sizes; (void)n_in; (void)out_size; (void)ws_size;
    const float* node  = (const float*)d_in[0];   // (4,64,128)
    const float* edge  = (const float*)d_in[1];   // (4,64,24,128)
    const float* W2    = (const float*)d_in[2];   // (128,256)
    const float* b2    = (const float*)d_in[3];   // (256)
    const float* W3    = (const float*)d_in[4];   // (640,256)
    const float* b3    = (const float*)d_in[5];   // (256)
    const float* gamma = (const float*)d_in[6];   // (128)
    const float* beta  = (const float*)d_in[7];   // (128)
    const float* nbr_mask = (const float*)d_in[8];// (4,64,24)
    const int*   nbr_idx  = (const int*)d_in[9];  // (4,64,24) int32
    float* out = (float*)d_out;

    float* ws = (float*)d_ws;
    float* PnA   = ws;                               // 256*256
    float* PnB   = PnA   + (size_t)NODES * 256;
    float* PnC   = PnB   + (size_t)NODES * 256;
    float* PeD   = PnC   + (size_t)NODES * 256;      // 6144*256
    float* PeE   = PeD   + (size_t)ROWS2 * 256;
    float* two_b = PeE   + (size_t)ROWS2 * 256;      // 6144*128
    float* three = two_b + (size_t)ROWS2 * FE_;      // 6144*128
    float* sums  = three + (size_t)ROWS2 * FE_;      // 256

    // Stage 1: projections through W3 row-slices (f16 WMMA GEMMs).
    gemm_proj_kernel<<<NODES / 16, 256, 0, stream>>>(node, W3 + (size_t)0   * 256, PnA);
    gemm_proj_kernel<<<NODES / 16, 256, 0, stream>>>(node, W3 + (size_t)128 * 256, PnB);
    gemm_proj_kernel<<<NODES / 16, 256, 0, stream>>>(node, W3 + (size_t)256 * 256, PnC);
    gemm_proj_kernel<<<ROWS2 / 16, 256, 0, stream>>>(edge, W3 + (size_t)384 * 256, PeD);
    gemm_proj_kernel<<<ROWS2 / 16, 256, 0, stream>>>(edge, W3 + (size_t)512 * 256, PeE);

    // Stage 2: two-body gated GEMM.
    twobody_kernel<<<ROWS2 / 16, 256, 0, stream>>>(node, W2, b2, nbr_idx, nbr_mask, two_b);

    // Stage 3: three-body gather-accumulate over k.
    threebody_kernel<<<ROWS2 / 8, 256, 0, stream>>>(PnA, PnB, PnC, PeD, PeE, b3,
                                                    nbr_idx, nbr_mask, three);

    // Stage 4: BN stats + fused epilogue.
    zero_sums_kernel<<<1, 256, 0, stream>>>(sums);
    bnsum_kernel<<<48, 256, 0, stream>>>(three, sums);
    final_kernel<<<(ROWS2 * FE_ + 255) / 256, 256, 0, stream>>>(
        edge, two_b, three, sums, gamma, beta, out);
}
